// AugmentedModel_23270132810349
// MI455X (gfx1250) — matrix operand order repaired and checked by reading, not compiled, not run on hardware
//
#include <hip/hip_runtime.h>
#include <math.h>

// MI455X (gfx1250, wave32). HMM forward restructured into linear-space with
// per-step renormalization so the transition update is a bf16 WMMA GEMM:
//   c_next[16 rows x 512] = c_h[16 x 512] @ T_a[512 x 512]   (f32 accumulate)
// T matrices (16 x 512 x 512 bf16 = 8MB) are prepacked in WMMA fragment order
// and stay resident in the 192MB L2. State lives in LDS for all 128 steps.

typedef __attribute__((ext_vector_type(16))) __bf16 v16bf;
typedef __attribute__((ext_vector_type(8)))  float  v8f;

#define S_DIM 512
#define B_DIM 64
#define T_DIM 128
#define A_DIM 16
#define O_DIM 4096
#define LOG2PI_F 1.8378770664093453f

__device__ __forceinline__ float wave_max(float v) {
#pragma unroll
  for (int o = 16; o > 0; o >>= 1) v = fmaxf(v, __shfl_xor(v, o, 32));
  return v;
}
__device__ __forceinline__ float wave_sum(float v) {
#pragma unroll
  for (int o = 16; o > 0; o >>= 1) v += __shfl_xor(v, o, 32);
  return v;
}

// ---- prep: prior softmax (512 values, one wave) ----
__global__ __launch_bounds__(32) void prep_prior_k(const float* __restrict__ pr,
                                                   float* __restrict__ prior) {
  int lane = threadIdx.x;
  float v[16];
  float mx = -INFINITY;
#pragma unroll
  for (int i = 0; i < 16; ++i) { v[i] = pr[lane + 32 * i]; mx = fmaxf(mx, v[i]); }
  mx = wave_max(mx);
  float sm = 0.f;
#pragma unroll
  for (int i = 0; i < 16; ++i) { v[i] = expf(v[i] - mx); sm += v[i]; }
  sm = wave_sum(sm);
  float inv = 1.0f / sm;
#pragma unroll
  for (int i = 0; i < 16; ++i) prior[lane + 32 * i] = v[i] * inv;
}

// ---- prep: log_softmax(obs_logits) stored TRANSPOSED [O][S] so the per-step
// gather of column o becomes a contiguous, coalesced 512-float read ----
__global__ __launch_bounds__(32) void prep_obs_k(const float* __restrict__ ol,
                                                 float* __restrict__ lobsT) {
  int s = blockIdx.x;
  int lane = threadIdx.x;
  const float* src = ol + (size_t)s * O_DIM;
  float mx = -INFINITY;
  for (int i = lane; i < O_DIM; i += 32) mx = fmaxf(mx, src[i]);
  mx = wave_max(mx);
  float sm = 0.f;
  for (int i = lane; i < O_DIM; i += 32) sm += expf(src[i] - mx);
  sm = wave_sum(sm);
  float lz = mx + logf(sm);
  for (int i = lane; i < O_DIM; i += 32) lobsT[(size_t)i * S_DIM + s] = src[i] - lz;
}

// ---- prep: log_softmax(policy_logits) over A ----
__global__ __launch_bounds__(512) void prep_pol_k(const float* __restrict__ pl,
                                                  float* __restrict__ lpol) {
  int s = blockIdx.x * blockDim.x + threadIdx.x;
  if (s >= S_DIM) return;
  const float* src = pl + (size_t)s * A_DIM;
  float mx = -INFINITY;
#pragma unroll
  for (int a = 0; a < A_DIM; ++a) mx = fmaxf(mx, src[a]);
  float sm = 0.f;
#pragma unroll
  for (int a = 0; a < A_DIM; ++a) sm += expf(src[a] - mx);
  float lz = mx + logf(sm);
#pragma unroll
  for (int a = 0; a < A_DIM; ++a) lpol[(size_t)s * A_DIM + a] = src[a] - lz;
}

// ---- prep: softmax(trans_logits, axis=-1) -> bf16, packed in B-fragment order.
// Block (a, kt, nt) = 32 lanes x 16 bf16 (1KB). Lane = n_in + 16*(k_in>>4),
// element e = k_in & 15 (contiguous k order, same as the A fragments).
__global__ __launch_bounds__(32) void prep_trans_k(const float* __restrict__ tl,
                                                   __bf16* __restrict__ Tb) {
  int row = blockIdx.x;            // a*512 + k
  int a = row >> 9, k = row & 511;
  int lane = threadIdx.x;
  const float* src = tl + (size_t)row * S_DIM;
  float v[16];
  float mx = -INFINITY;
#pragma unroll
  for (int i = 0; i < 16; ++i) { v[i] = src[lane + 32 * i]; mx = fmaxf(mx, v[i]); }
  mx = wave_max(mx);
  float sm = 0.f;
#pragma unroll
  for (int i = 0; i < 16; ++i) { v[i] = expf(v[i] - mx); sm += v[i]; }
  sm = wave_sum(sm);
  float inv = 1.0f / sm;
  int kt = k >> 5, k_in = k & 31;
  int e = k_in & 15;
  int laneHalf = (k_in & 16);      // 0 or 16
#pragma unroll
  for (int i = 0; i < 16; ++i) {
    int n = lane + 32 * i;
    int nt = n >> 4, n_in = n & 15;
    size_t blk = (size_t)(a * 16 + kt) * 32 + nt;
    Tb[blk * 512 + (size_t)(n_in + laneHalf) * 16 + e] = (__bf16)(v[i] * inv);
  }
}

// ---- main: 4 WGs x 512 threads (16 waves). Wave m owns batch row m in the
// emission phase; in the matmul phase each wave owns N-tiles {w, w+16}. ----
__global__ __launch_bounds__(512)
void hmm_forward_k(const float* __restrict__ regime, const int* __restrict__ obs,
                   const float* __restrict__ rewards, const float* __restrict__ dones,
                   const int* __restrict__ actions, const float* __restrict__ r_mu,
                   const float* __restrict__ r_logsig, const __bf16* __restrict__ Tb,
                   const float* __restrict__ lobsT, const float* __restrict__ lpol,
                   const float* __restrict__ prior, float* __restrict__ out) {
  __shared__ float  cbuf[16][S_DIM];   // fp32 chain state (32KB)
  __shared__ __bf16 cbf[16][S_DIM];    // bf16 copy for WMMA A-frags (16KB)
  __shared__ float  base_s[16], logp_s[16], wd_s[16];
  __shared__ int    act_s[16];

  const int tid  = threadIdx.x;
  const int lane = tid & 31;
  const int wv   = tid >> 5;               // wave id == row m
  const int b    = blockIdx.x * 16 + wv;   // global batch index
  const float reg = regime[b];

#pragma unroll
  for (int i = 0; i < 16; ++i) cbuf[wv][lane + 32 * i] = prior[lane + 32 * i];
  if (lane == 0) { base_s[wv] = 0.f; logp_s[wv] = 0.f; wd_s[wv] = 0.f; }
  __syncthreads();

  for (int t = 0; t <= T_DIM; ++t) {
    // ---------------- emission + renormalize (wave wv -> row wv) ----------
    const bool last = (t == T_DIM);
    const int   o = obs[t * B_DIM + b];
    const float r = rewards[t * B_DIM + b];
    const float d = dones[t * B_DIM + b];
    const int   a = last ? 0 : actions[t * B_DIM + b];
    const float wd   = wd_s[wv];
    const float done = last ? wd : fmaxf(wd, d);
    const bool  mask_act = last || (done == 1.0f) || (reg == 1.0f);
    const float* lob = lobsT + (size_t)o * S_DIM;

    float ev[16], cv[16];
    float mx = -INFINITY;
#pragma unroll
    for (int i = 0; i < 16; ++i) {
      int s = lane + 32 * i;
      float mu = r_mu[s], ls = r_logsig[s];
      float tdf = (r - mu) * expf(-ls);
      float e = lob[s] - 0.5f * tdf * tdf - ls - 0.5f * LOG2PI_F;
      if (!mask_act) e += lpol[(size_t)s * A_DIM + a];
      ev[i] = e;
      cv[i] = cbuf[wv][s];
      mx = fmaxf(mx, e);
    }
    mx = wave_max(mx);
    float sm = 0.f;
#pragma unroll
    for (int i = 0; i < 16; ++i) { cv[i] *= expf(ev[i] - mx); sm += cv[i]; }
    sm = wave_sum(sm);
    float lse_lq  = base_s[wv] + mx + logf(sm);
    float lq_orda = lse_lq * (1.0f - wd);
    float inv = 1.0f / sm;
#pragma unroll
    for (int i = 0; i < 16; ++i) {
      int s = lane + 32 * i;
      float ch = cv[i] * inv;
      cbuf[wv][s] = ch;
      cbf[wv][s]  = (__bf16)ch;
    }
    if (lane == 0) {
      float lp = logp_s[wv];
      if (!isinf(lp)) lp += lq_orda;
      logp_s[wv] = lp;
      base_s[wv] = lse_lq - lq_orda;
      wd_s[wv]   = done;
      act_s[wv]  = a;
    }
    __syncthreads();
    if (last) break;

    // ---------------- transition: C_a = c_h @ T_a, keep rows with act==a ---
    unsigned amask = 0;
#pragma unroll
    for (int m = 0; m < 16; ++m) amask |= 1u << act_s[m];

    const int half = lane >> 4;
    const int l15  = lane & 15;
    const int nt0 = wv, nt1 = wv + 16;

    // Register-cache the action id of the 8 C-tile rows this lane owns so the
    // selection below is pure VALU (v_cndmask), no per-action LDS traffic.
    int actm[8];
#pragma unroll
    for (int v = 0; v < 8; ++v) actm[v] = act_s[v + 8 * half];

    v8f res0 = {}; v8f res1 = {};
    for (int a2 = 0; a2 < A_DIM; ++a2) {
      if (!((amask >> a2) & 1)) continue;        // WG-uniform branch
      v8f acc0 = {}; v8f acc1 = {};
      const __bf16* TbA = Tb + (size_t)a2 * 16 * 32 * 512;
#pragma unroll 4
      for (int kt = 0; kt < 16; ++kt) {
        // A fragment (16x32 bf16): lane = half*16 + m; elems 0-15 -> k local
        // [16*half, 16*half+15].
        v16bf af;
        {
          const __bf16* arow = &cbf[l15][kt * 32];
          uint4* ap = reinterpret_cast<uint4*>(&af);
          ap[0] = *reinterpret_cast<const uint4*>(arow + 16 * half);
          ap[1] = *reinterpret_cast<const uint4*>(arow + 16 * half + 8);
        }
        // B fragments: prepacked — lane reads its own contiguous 32 bytes.
        const __bf16* bbase = TbA + (size_t)kt * 32 * 512;
        v16bf bf0, bf1;
        {
          const uint4* bp0 = reinterpret_cast<const uint4*>(bbase + (size_t)nt0 * 512 + lane * 16);
          uint4* q0 = reinterpret_cast<uint4*>(&bf0);
          q0[0] = bp0[0]; q0[1] = bp0[1];
          const uint4* bp1 = reinterpret_cast<const uint4*>(bbase + (size_t)nt1 * 512 + lane * 16);
          uint4* q1 = reinterpret_cast<uint4*>(&bf1);
          q1[0] = bp1[0]; q1[1] = bp1[1];
        }
        acc0 = __builtin_amdgcn_wmma_f32_16x16x32_bf16(false, af, false, bf0,
                                                       (short)0, acc0, false, false);
        acc1 = __builtin_amdgcn_wmma_f32_16x16x32_bf16(false, af, false, bf1,
                                                       (short)0, acc1, false, false);
      }
      // Select matching rows into result registers (pure VALU, no LDS).
#pragma unroll
      for (int v = 0; v < 8; ++v) {
        res0[v] = (actm[v] == a2) ? acc0[v] : res0[v];
        res1[v] = (actm[v] == a2) ? acc1[v] : res1[v];
      }
    }
    // Single unconditional writeback: row m = v + 8*half, col = nt*16 + l15.
#pragma unroll
    for (int v = 0; v < 8; ++v) {
      int m = v + 8 * half;
      cbuf[m][nt0 * 16 + l15] = res0[v];
      cbuf[m][nt1 * 16 + l15] = res1[v];
    }
    __syncthreads();
  }

  if (lane == 0) out[b] = logp_s[wv];
}

extern "C" void kernel_launch(void* const* d_in, const int* in_sizes, int n_in,
                              void* d_out, int out_size, void* d_ws, size_t ws_size,
                              hipStream_t stream) {
  (void)in_sizes; (void)n_in; (void)out_size; (void)ws_size;
  const float* regime        = (const float*)d_in[0];
  const int*   obs           = (const int*)d_in[1];
  const float* rewards       = (const float*)d_in[2];
  const float* dones         = (const float*)d_in[3];
  const int*   actions       = (const int*)d_in[4];
  const float* prior_logits  = (const float*)d_in[5];
  const float* trans_logits  = (const float*)d_in[6];
  const float* obs_logits    = (const float*)d_in[7];
  const float* policy_logits = (const float*)d_in[8];
  const float* r_mu          = (const float*)d_in[9];
  const float* r_logsig      = (const float*)d_in[10];

  char* ws = (char*)d_ws;
  __bf16* Tb   = (__bf16*)ws;                                // 16*512*512*2 = 8 MB
  float* lobsT = (float*)(ws + (size_t)(8u << 20));          // 4096*512*4  = 8 MB
  float* lpol  = (float*)(ws + (size_t)(16u << 20));         // 512*16*4    = 32 KB
  float* prior = (float*)(ws + (size_t)(16u << 20) + 32768); // 512*4       = 2 KB

  hipLaunchKernelGGL(prep_trans_k, dim3(A_DIM * S_DIM), dim3(32), 0, stream,
                     trans_logits, Tb);
  hipLaunchKernelGGL(prep_obs_k, dim3(S_DIM), dim3(32), 0, stream,
                     obs_logits, lobsT);
  hipLaunchKernelGGL(prep_pol_k, dim3(1), dim3(512), 0, stream,
                     policy_logits, lpol);
  hipLaunchKernelGGL(prep_prior_k, dim3(1), dim3(32), 0, stream,
                     prior_logits, prior);
  hipLaunchKernelGGL(hmm_forward_k, dim3(B_DIM / 16), dim3(512), 0, stream,
                     regime, obs, rewards, dones, actions, r_mu, r_logsig,
                     Tb, lobsT, lpol, prior, (float*)d_out);
}
